// LSTMbyHand_44547400794614
// MI455X (gfx1250) — matrix operand order repaired and checked by reading, not compiled, not run on hardware
//
#include <hip/hip_runtime.h>

typedef __bf16 bf16_t;
typedef __attribute__((ext_vector_type(16))) __bf16 v16bf;
typedef __attribute__((ext_vector_type(8)))  __bf16 v8bf;
typedef __attribute__((ext_vector_type(8)))  float  v8f;

#define LSTM_L 2
#define LSTM_B 64
#define LSTM_S 512
#define LSTM_H 1024
#define LSTM_I 1024
#define GATES  4
#define NCOL   (GATES * LSTM_H)   // 4096

// ---------------------------------------------------------------------------
// fp32 -> bf16 elementwise convert (grid-stride)
// ---------------------------------------------------------------------------
__global__ void cvt_f32_bf16(const float* __restrict__ src,
                             bf16_t* __restrict__ dst, int n) {
  int i = blockIdx.x * blockDim.x + threadIdx.x;
  int stride = gridDim.x * blockDim.x;
  for (; i < n; i += stride) dst[i] = (bf16_t)src[i];
}

// bsum = bi + bh (per layer, 4H each)
__global__ void bias_sum(const float* __restrict__ bi,
                         const float* __restrict__ bh,
                         float* __restrict__ bsum, int n) {
  int i = blockIdx.x * blockDim.x + threadIdx.x;
  int stride = gridDim.x * blockDim.x;
  for (; i < n; i += stride) bsum[i] = bi[i] + bh[i];
}

// h/c state init: state lives in the hn/cn regions of d_out
__global__ void init_state(const float* __restrict__ h0,
                           const float* __restrict__ c0,
                           float* __restrict__ hstate,
                           float* __restrict__ cstate,
                           bf16_t* __restrict__ hbf, int n) {
  int i = blockIdx.x * blockDim.x + threadIdx.x;
  int stride = gridDim.x * blockDim.x;
  for (; i < n; i += stride) {
    float h = h0[i];
    hstate[i] = h;
    cstate[i] = c0[i];
    hbf[i] = (bf16_t)h;
  }
}

// ---------------------------------------------------------------------------
// Fused dual GEMM: z[64,4096] = A1[64,1024] @ W1[4096,1024]^T
//                             + A2[64,1024] @ W2[4096,1024]^T
// All operands bf16 (K-contiguous), fp32 accumulate via v_wmma_f32_16x16x32_bf16.
// Block = 128 threads (4 waves), each wave owns a 16(M) x 64(N) tile.
// Grid.x = 4096/64 = 64 blocks.
// ---------------------------------------------------------------------------
__device__ __forceinline__ v16bf load_a_frag(const bf16_t* __restrict__ row,
                                             int half) {
  // A fragment (16-bit, 16x32): lane<16 holds K {0..15}, lane>=16 holds
  // K {16..31} of its row -> one 32B contiguous load.
  return *(const v16bf*)(row + half * 16);
}

__global__ __launch_bounds__(128) void lstm_gemm(
    const bf16_t* __restrict__ A1, const bf16_t* __restrict__ W1,
    const bf16_t* __restrict__ A2, const bf16_t* __restrict__ W2,
    float* __restrict__ z) {
  const int K = 1024;
  const int lane = threadIdx.x & 31;
  const int wave = threadIdx.x >> 5;
  const int half = lane >> 4;   // 0 / 1
  const int lr   = lane & 15;
  const int rowBase = wave * 16;          // M tile: 0,16,32,48
  const int n0 = blockIdx.x * 64;         // N tile base

  v8f zero = {};
  v8f acc[4] = {zero, zero, zero, zero};

  const bf16_t* a1row = A1 + (size_t)(rowBase + lr) * K;
  const bf16_t* a2row = A2 + (size_t)(rowBase + lr) * K;

  // Pass 1: A1 x W1^T
  for (int k0 = 0; k0 < K; k0 += 32) {
    v16bf af = load_a_frag(a1row + k0, half);
#pragma unroll
    for (int j = 0; j < 4; ++j) {
      // B fragment (32x16): lane holds output column (n0+j*16+lr),
      // 16 contiguous K starting at k0 + half*16 -> one 32B load.
      v16bf bfg = *(const v16bf*)(W1 + (size_t)(n0 + j * 16 + lr) * K +
                                  k0 + half * 16);
      acc[j] = __builtin_amdgcn_wmma_f32_16x16x32_bf16(
          false, af, false, bfg, (short)0, acc[j], false, false);
    }
  }
  // Pass 2: A2 x W2^T (accumulates into same tiles)
  for (int k0 = 0; k0 < K; k0 += 32) {
    v16bf af = load_a_frag(a2row + k0, half);
#pragma unroll
    for (int j = 0; j < 4; ++j) {
      v16bf bfg = *(const v16bf*)(W2 + (size_t)(n0 + j * 16 + lr) * K +
                                  k0 + half * 16);
      acc[j] = __builtin_amdgcn_wmma_f32_16x16x32_bf16(
          false, af, false, bfg, (short)0, acc[j], false, false);
    }
  }

  // C tile layout: lanes 0-15 -> N=lr, M=v ; lanes 16-31 -> N=lr, M=8+v
#pragma unroll
  for (int j = 0; j < 4; ++j) {
#pragma unroll
    for (int v = 0; v < 8; ++v) {
      z[(size_t)(rowBase + v + 8 * half) * NCOL + n0 + j * 16 + lr] = acc[j][v];
    }
  }
}

// ---------------------------------------------------------------------------
// Gate nonlinearity: one block per batch row b.
// z row is 4096 wide: [i | f | g | o]; softmax over H=1024 for i,f,o; tanh g.
// cl = f*c + i*tanh(zg);  hl = o*tanh(tanh(zg)).
// ---------------------------------------------------------------------------
__device__ __forceinline__ float block_max(float v, float* sm) {
  int t = threadIdx.x;
  sm[t] = v;
  __syncthreads();
  for (int s = 128; s > 0; s >>= 1) {
    if (t < s) sm[t] = fmaxf(sm[t], sm[t + s]);
    __syncthreads();
  }
  float r = sm[0];
  __syncthreads();
  return r;
}

__device__ __forceinline__ float block_sum(float v, float* sm) {
  int t = threadIdx.x;
  sm[t] = v;
  __syncthreads();
  for (int s = 128; s > 0; s >>= 1) {
    if (t < s) sm[t] += sm[t + s];
    __syncthreads();
  }
  float r = sm[0];
  __syncthreads();
  return r;
}

__global__ __launch_bounds__(256) void lstm_gate(
    const float* __restrict__ z, const float* __restrict__ bsum,
    float* __restrict__ c_state, float* __restrict__ h_state,
    bf16_t* __restrict__ h_bf, float* __restrict__ seq_out) {
  __shared__ float sm[256];
  const int b = blockIdx.x;
  const int t = threadIdx.x;
  const float* zr = z + (size_t)b * NCOL;

  float zi[4], zf[4], zg[4], zo[4];
#pragma unroll
  for (int j = 0; j < 4; ++j) {
    int h = t + 256 * j;
    zi[j] = zr[h]        + bsum[h];
    zf[j] = zr[1024 + h] + bsum[1024 + h];
    zg[j] = zr[2048 + h] + bsum[2048 + h];
    zo[j] = zr[3072 + h] + bsum[3072 + h];
  }

  float iv[4], fv[4], ov[4];
  // softmax(i)
  {
    float m = fmaxf(fmaxf(zi[0], zi[1]), fmaxf(zi[2], zi[3]));
    m = block_max(m, sm);
    float s = 0.f;
#pragma unroll
    for (int j = 0; j < 4; ++j) { iv[j] = expf(zi[j] - m); s += iv[j]; }
    s = block_sum(s, sm);
    float inv = 1.0f / s;
#pragma unroll
    for (int j = 0; j < 4; ++j) iv[j] *= inv;
  }
  // softmax(f)
  {
    float m = fmaxf(fmaxf(zf[0], zf[1]), fmaxf(zf[2], zf[3]));
    m = block_max(m, sm);
    float s = 0.f;
#pragma unroll
    for (int j = 0; j < 4; ++j) { fv[j] = expf(zf[j] - m); s += fv[j]; }
    s = block_sum(s, sm);
    float inv = 1.0f / s;
#pragma unroll
    for (int j = 0; j < 4; ++j) fv[j] *= inv;
  }
  // softmax(o)
  {
    float m = fmaxf(fmaxf(zo[0], zo[1]), fmaxf(zo[2], zo[3]));
    m = block_max(m, sm);
    float s = 0.f;
#pragma unroll
    for (int j = 0; j < 4; ++j) { ov[j] = expf(zo[j] - m); s += ov[j]; }
    s = block_sum(s, sm);
    float inv = 1.0f / s;
#pragma unroll
    for (int j = 0; j < 4; ++j) ov[j] *= inv;
  }

#pragma unroll
  for (int j = 0; j < 4; ++j) {
    int h = t + 256 * j;
    size_t idx = (size_t)b * LSTM_H + h;
    float g  = tanhf(zg[j]);
    float cn = fv[j] * c_state[idx] + iv[j] * g;
    float hn = ov[j] * tanhf(g);
    c_state[idx] = cn;
    h_state[idx] = hn;
    h_bf[idx] = (bf16_t)hn;
    if (seq_out) seq_out[idx] = hn;
  }
}

// ---------------------------------------------------------------------------
extern "C" void kernel_launch(void* const* d_in, const int* in_sizes, int n_in,
                              void* d_out, int out_size, void* d_ws,
                              size_t ws_size, hipStream_t stream) {
  (void)in_sizes; (void)n_in; (void)out_size; (void)ws_size;
  const float* x  = (const float*)d_in[0];  // [S,B,I]
  const float* h0 = (const float*)d_in[1];  // [L,B,H]
  const float* c0 = (const float*)d_in[2];  // [L,B,H]
  const float* Wi = (const float*)d_in[3];  // [L,4,H,I]
  const float* Wh = (const float*)d_in[4];  // [L,4,H,H]
  const float* bi = (const float*)d_in[5];  // [L,4,H]
  const float* bh = (const float*)d_in[6];  // [L,4,H]
  float* out = (float*)d_out;

  const size_t WELEMS  = (size_t)LSTM_L * GATES * LSTM_H * LSTM_I; // per-tensor
  const size_t WSTRIDE = (size_t)GATES * LSTM_H * LSTM_I;          // per-layer

  char* ws = (char*)d_ws;
  size_t off = 0;
  bf16_t* Wi_bf = (bf16_t*)(ws + off); off += WELEMS * 2;                      // 16.8 MB
  bf16_t* Wh_bf = (bf16_t*)(ws + off); off += WELEMS * 2;                      // 16.8 MB
  bf16_t* x_bf  = (bf16_t*)(ws + off); off += (size_t)LSTM_B * LSTM_I * 2;     // 128 KB
  bf16_t* h_bf  = (bf16_t*)(ws + off); off += (size_t)LSTM_L * LSTM_B * LSTM_H * 2;
  float*  zbuf  = (float*)(ws + off);  off += (size_t)LSTM_B * NCOL * 4;       // 1 MB
  float*  bsum  = (float*)(ws + off);  off += (size_t)LSTM_L * NCOL * 4;

  // Output layout: seq [S,B,H] | hn [L,B,H] | cn [L,B,H].
  // h/c running state lives directly in the hn/cn regions.
  float* seq = out;
  float* hn  = out + (size_t)LSTM_S * LSTM_B * LSTM_H;
  float* cn  = hn + (size_t)LSTM_L * LSTM_B * LSTM_H;

  // One-time prep (re-done every launch for determinism)
  cvt_f32_bf16<<<4096, 256, 0, stream>>>(Wi, Wi_bf, (int)WELEMS);
  cvt_f32_bf16<<<4096, 256, 0, stream>>>(Wh, Wh_bf, (int)WELEMS);
  bias_sum<<<32, 256, 0, stream>>>(bi, bh, bsum, LSTM_L * NCOL);
  init_state<<<512, 256, 0, stream>>>(h0, c0, hn, cn, h_bf,
                                      LSTM_L * LSTM_B * LSTM_H);

  bf16_t* hbf0 = h_bf;
  bf16_t* hbf1 = h_bf + (size_t)LSTM_B * LSTM_H;
  float* c0s = cn;
  float* c1s = cn + (size_t)LSTM_B * LSTM_H;
  float* h0s = hn;
  float* h1s = hn + (size_t)LSTM_B * LSTM_H;

  for (int t = 0; t < LSTM_S; ++t) {
    // layer 0
    cvt_f32_bf16<<<64, 256, 0, stream>>>(x + (size_t)t * LSTM_B * LSTM_I,
                                         x_bf, LSTM_B * LSTM_I);
    lstm_gemm<<<64, 128, 0, stream>>>(x_bf, Wi_bf, hbf0, Wh_bf, zbuf);
    lstm_gate<<<64, 256, 0, stream>>>(zbuf, bsum, c0s, h0s, hbf0, nullptr);
    // layer 1 (input = fresh h of layer 0)
    lstm_gemm<<<64, 128, 0, stream>>>(hbf0, Wi_bf + WSTRIDE, hbf1,
                                      Wh_bf + WSTRIDE, zbuf);
    lstm_gate<<<64, 256, 0, stream>>>(zbuf, bsum + NCOL, c1s, h1s, hbf1,
                                      seq + (size_t)t * LSTM_B * LSTM_H);
  }
}
